// DisentangledSelfAttention_78726750536334
// MI455X (gfx1250) — matrix operand order repaired and checked by reading, not compiled, not run on hardware
//
#include <hip/hip_runtime.h>
#include <hip/hip_bf16.h>
#include <math.h>

// DeBERTa disentangled self-attention, algebraically refactored:
//   c2p[h,i,j] = pos[i,j,:] . Qt[i,h,:]           (Qt = q @ Wpk_h^T)
//   p2c[h,i,j] = pos[j,i,:] . Kt[h,j,:] + ck[h,j] (Kt = k @ Wpq_h^T)
// Single streaming pass over pos_emb (453 MB) computes both -> memory bound
// (~20us @ 23.3 TB/s). All GEMMs via v_wmma_f32_16x16x32_bf16 (wave32).

typedef __bf16 bf16_t;
typedef bf16_t v16bf __attribute__((ext_vector_type(16)));
typedef float  v8f   __attribute__((ext_vector_type(8)));
typedef float  v4f   __attribute__((ext_vector_type(4)));

#define SEQ   384
#define HID   768
#define NHEAD 12
#define DHEAD 64

__device__ __forceinline__ v8f wmma_bf16(v16bf a, v16bf b, v8f c) {
  return __builtin_amdgcn_wmma_f32_16x16x32_bf16(false, a, false, b, (short)0, c,
                                                 false, false);
}

// A operand (16x32 bf16) from row-major f32, non-temporal (streaming).
// Layout (ISA 7.12.2): lane<16 -> K in {0..7,16..23}; lane>=16 -> {8..15,24..31}.
__device__ __forceinline__ v16bf load_A_f32_nt(const float* base, size_t ld,
                                               int r0, int c0) {
  const int lane = threadIdx.x & 31;
  const int m    = lane & 15;
  const int kb   = (lane & 16) ? 8 : 0;
  const float* p = base + (size_t)(r0 + m) * ld + c0 + kb;
  v4f x0 = __builtin_nontemporal_load((const v4f*)p);
  v4f x1 = __builtin_nontemporal_load((const v4f*)(p + 4));
  v4f x2 = __builtin_nontemporal_load((const v4f*)(p + 16));
  v4f x3 = __builtin_nontemporal_load((const v4f*)(p + 20));
  v16bf a;
#pragma unroll
  for (int e = 0; e < 4; ++e) {
    a[e]      = (bf16_t)x0[e];
    a[4 + e]  = (bf16_t)x1[e];
    a[8 + e]  = (bf16_t)x2[e];
    a[12 + e] = (bf16_t)x3[e];
  }
  return a;
}

// A operand from row-major f32 (cached path).
__device__ __forceinline__ v16bf load_A_f32(const float* base, size_t ld,
                                            int r0, int c0) {
  const int lane = threadIdx.x & 31;
  const int m    = lane & 15;
  const int kb   = (lane & 16) ? 8 : 0;
  const float* p = base + (size_t)(r0 + m) * ld + c0 + kb;
  v4f x0 = *(const v4f*)p;
  v4f x1 = *(const v4f*)(p + 4);
  v4f x2 = *(const v4f*)(p + 16);
  v4f x3 = *(const v4f*)(p + 20);
  v16bf a;
#pragma unroll
  for (int e = 0; e < 4; ++e) {
    a[e]      = (bf16_t)x0[e];
    a[4 + e]  = (bf16_t)x1[e];
    a[8 + e]  = (bf16_t)x2[e];
    a[12 + e] = (bf16_t)x3[e];
  }
  return a;
}

// A operand from row-major bf16.
__device__ __forceinline__ v16bf load_A_bf16(const bf16_t* base, size_t ld,
                                             int r0, int c0) {
  const int lane = threadIdx.x & 31;
  const int m    = lane & 15;
  const int kb   = (lane & 16) ? 8 : 0;
  const bf16_t* p = base + (size_t)(r0 + m) * ld + c0 + kb;
  v16bf a;
#pragma unroll
  for (int e = 0; e < 8; ++e) {
    a[e]     = p[e];
    a[8 + e] = p[16 + e];
  }
  return a;
}

// B operand (32x16 bf16) from N-major storage: element(K=k,N=n) = base[n*ld + k].
// Layout: lanes 0-15 -> N=lane, K={0..7,16..23}; lanes 16-31 -> N=lane-16, K={8..15,24..31}.
__device__ __forceinline__ v16bf load_B_bf16(const bf16_t* base, size_t ld,
                                             int n0, int c0) {
  const int lane = threadIdx.x & 31;
  const int n    = n0 + (lane & 15);
  const int kb   = (lane & 16) ? 8 : 0;
  const bf16_t* p = base + (size_t)n * ld + c0 + kb;
  v16bf b;
#pragma unroll
  for (int e = 0; e < 8; ++e) {
    b[e]     = p[e];
    b[8 + e] = p[16 + e];
  }
  return b;
}

// ---------- Kernel A0: convert / transpose weights to bf16 ----------
__global__ __launch_bounds__(256) void convert_weights_kernel(
    const float* __restrict__ Wq, const float* __restrict__ Wk,
    const float* __restrict__ Wv, const float* __restrict__ Wpk,
    const float* __restrict__ Wpq, bf16_t* __restrict__ WTq,
    bf16_t* __restrict__ WTk, bf16_t* __restrict__ WTv,
    bf16_t* __restrict__ WPk, bf16_t* __restrict__ WPq) {
  int idx = blockIdx.x * blockDim.x + threadIdx.x;
  if (idx >= HID * HID) return;
  int i = idx / HID;  // input dim
  int o = idx % HID;  // output dim
  WTq[(size_t)o * HID + i] = (bf16_t)Wq[idx];
  WTk[(size_t)o * HID + i] = (bf16_t)Wk[idx];
  WTv[(size_t)o * HID + i] = (bf16_t)Wv[idx];
  WPk[idx] = (bf16_t)Wpk[idx];
  WPq[idx] = (bf16_t)Wpq[idx];
}

// ---------- Kernel A1: QKV projections (M=384, N=768, K=768 per matrix) ----
__global__ __launch_bounds__(256) void qkv_kernel(
    const float* __restrict__ hs, const bf16_t* __restrict__ WT,
    const float* __restrict__ bq, const float* __restrict__ bk,
    const float* __restrict__ bv, bf16_t* __restrict__ qb,
    bf16_t* __restrict__ kb, bf16_t* __restrict__ vT) {
  const int it   = blockIdx.x;       // 24 i-tiles
  const int mat  = blockIdx.y;       // 0=q 1=k 2=v
  const int wave = threadIdx.x >> 5;
  const int lane = threadIdx.x & 31;
  const bf16_t* W = WT + (size_t)mat * HID * HID;
  const float* bias = (mat == 0) ? bq : ((mat == 1) ? bk : bv);
  const int i0 = it * 16;
  for (int nt = wave; nt < 48; nt += 8) {
    v8f acc = {};
    for (int ks = 0; ks < 24; ++ks) {
      v16bf a = load_A_f32(hs, HID, i0, ks * 32);
      v16bf b = load_B_bf16(W, HID, nt * 16, ks * 32);
      acc = wmma_bf16(a, b, acc);
    }
    const int ng = nt * 16 + (lane & 15);
    const int h = ng >> 6, d = ng & 63;
    const int mh = (lane & 16) ? 8 : 0;
    const float bi = bias[ng];
    // Branch-free store loop: select destination + stride once (mat uniform).
    bf16_t* dst;
    size_t stride;
    if (mat == 2) {  // v stored transposed: vT[h][d][i]
      dst = vT + ((size_t)h * DHEAD + d) * SEQ + (i0 + mh);
      stride = 1;
    } else {         // q/k stored [h][i][d]
      bf16_t* qk = (mat == 0) ? qb : kb;
      dst = qk + ((size_t)h * SEQ + (i0 + mh)) * DHEAD + d;
      stride = DHEAD;
    }
#pragma unroll
    for (int r = 0; r < 8; ++r) dst[(size_t)r * stride] = (bf16_t)(acc[r] + bi);
  }
}

// ---------- ck[h][j] = k[h,j,:] . bpq_h  (p2c bias term) ----------
__global__ __launch_bounds__(256) void ck_kernel(const bf16_t* __restrict__ kb,
                                                 const float* __restrict__ bpq,
                                                 float* __restrict__ ckb) {
  int idx = blockIdx.x * blockDim.x + threadIdx.x;
  if (idx >= NHEAD * SEQ) return;
  int h = idx / SEQ, j = idx % SEQ;
  float s = 0.f;
#pragma unroll 8
  for (int d = 0; d < DHEAD; ++d)
    s += (float)kb[((size_t)h * SEQ + j) * DHEAD + d] * bpq[h * DHEAD + d];
  ckb[idx] = s;
}

// ---------- Kernel A2: G[slot][a][c] = Qt (slot<12) / Kt (12..23) -----------
__global__ __launch_bounds__(256) void g_kernel(
    const bf16_t* __restrict__ qb, const bf16_t* __restrict__ kb,
    const bf16_t* __restrict__ WPk, const bf16_t* __restrict__ WPq,
    bf16_t* __restrict__ G) {
  const int mt   = blockIdx.x;   // 24 a-tiles
  const int slot = blockIdx.y;   // 24 slots
  const int h    = slot % NHEAD;
  const bool isQ = slot < NHEAD;
  const bf16_t* src = (isQ ? qb : kb) + (size_t)h * SEQ * DHEAD;
  const bf16_t* WP  = (isQ ? WPk : WPq) + h * DHEAD;  // column block of Wp
  const int wave = threadIdx.x >> 5;
  const int lane = threadIdx.x & 31;
  const int a0 = mt * 16;
  for (int nt = wave; nt < 48; nt += 8) {
    v8f acc = {};
    for (int ks = 0; ks < 2; ++ks) {
      v16bf a = load_A_bf16(src, DHEAD, a0, ks * 32);
      v16bf b = load_B_bf16(WP, HID, nt * 16, ks * 32);  // (K=d, N=c)=Wp[c][hD+d]
      acc = wmma_bf16(a, b, acc);
    }
    bf16_t* dst = G + ((size_t)slot * SEQ + a0) * HID + nt * 16 + (lane & 15);
    const int mh = (lane & 16) ? 8 : 0;
#pragma unroll
    for (int r = 0; r < 8; ++r) dst[(size_t)(r + mh) * HID] = (bf16_t)acc[r];
  }
}

// ---------- Kernel B: streaming pass over pos_emb (the 453 MB) --------------
// Block ai: (384_b x 768_c) @ G[ai]^T (768_c x 32_slots) -> pbuf[ai][slot][b]
//   slot<12 : c2p[h=slot, i=ai, j=b]
//   12..23  : p2c[h=slot-12, i=b, j=ai]
__global__ __launch_bounds__(256) void pos_kernel(
    const float* __restrict__ pos, const bf16_t* __restrict__ G,
    float* __restrict__ pbuf) {
  const int ai   = blockIdx.x;
  const int wave = threadIdx.x >> 5;
  const int lane = threadIdx.x & 31;
  const float*  pa = pos + (size_t)ai * SEQ * HID;
  const bf16_t* Ga = G + (size_t)ai * HID;  // slot stride = SEQ*HID
  float* outa = pbuf + (size_t)ai * 24 * SEQ;
  for (int bt = wave; bt < 24; bt += 8) {
    const int b0 = bt * 16;
    v8f acc0 = {}, acc1 = {};
    for (int ks = 0; ks < 24; ++ks) {
      const int c0 = ks * 32;
      v16bf a  = load_A_f32_nt(pa, HID, b0, c0);
      v16bf bA = load_B_bf16(Ga, (size_t)SEQ * HID, 0, c0);
      v16bf bB = load_B_bf16(Ga, (size_t)SEQ * HID, 16, c0);
      acc0 = wmma_bf16(a, bA, acc0);
      acc1 = wmma_bf16(a, bB, acc1);
    }
    const int bhi = b0 + ((lane & 16) ? 8 : 0);
#pragma unroll
    for (int half = 0; half < 2; ++half) {
      const v8f acc = half ? acc1 : acc0;
      const int slot = half * 16 + (lane & 15);
      if (slot >= 24) continue;  // padding slots 24-31 (lanes 8-15 of half 1)
      float* dst = outa + (size_t)slot * SEQ + bhi;
      v4f lo = {acc[0], acc[1], acc[2], acc[3]};
      v4f hi = {acc[4], acc[5], acc[6], acc[7]};
      *(v4f*)dst       = lo;
      *(v4f*)(dst + 4) = hi;
    }
  }
}

// ---------- Kernel C: c2c[h][i][j] = q . k ----------------------------------
__global__ __launch_bounds__(256) void c2c_kernel(
    const bf16_t* __restrict__ qb, const bf16_t* __restrict__ kb,
    float* __restrict__ c2c) {
  const int it = blockIdx.x;  // 24 i-tiles
  const int h  = blockIdx.y;  // 12 heads
  const int wave = threadIdx.x >> 5;
  const int lane = threadIdx.x & 31;
  const bf16_t* qh = qb + (size_t)h * SEQ * DHEAD;
  const bf16_t* kh = kb + (size_t)h * SEQ * DHEAD;
  const int i0 = it * 16;
  for (int jt = wave; jt < 24; jt += 8) {
    v8f acc = {};
    for (int ks = 0; ks < 2; ++ks) {
      v16bf a = load_A_bf16(qh, DHEAD, i0, ks * 32);
      v16bf b = load_B_bf16(kh, DHEAD, jt * 16, ks * 32);  // (K=d,N=j)=k[j][d]
      acc = wmma_bf16(a, b, acc);
    }
    float* dst = c2c + ((size_t)h * SEQ + i0) * SEQ + jt * 16 + (lane & 15);
    const int mh = (lane & 16) ? 8 : 0;
#pragma unroll
    for (int r = 0; r < 8; ++r) dst[(size_t)(r + mh) * SEQ] = acc[r];
  }
}

// ---------- Kernel D: assemble scores + softmax -> probs (bf16) -------------
__global__ __launch_bounds__(128) void softmax_kernel(
    const float* __restrict__ c2c, const float* __restrict__ pbuf,
    const float* __restrict__ ckb, const float* __restrict__ mask,
    bf16_t* __restrict__ probs) {
  const int h = blockIdx.x % NHEAD;
  const int i = blockIdx.x / NHEAD;
  const int t = threadIdx.x;
  __shared__ float red[128];
  float sv[3];
  float mx = -3.0e38f;
#pragma unroll
  for (int u = 0; u < 3; ++u) {
    const int j = t + 128 * u;
    float x = c2c[((size_t)h * SEQ + i) * SEQ + j]
            + pbuf[((size_t)i * 24 + h) * SEQ + j]              // c2p[h,i,j]
            + pbuf[((size_t)j * 24 + NHEAD + h) * SEQ + i]      // p2c[h,i,j]
            + ckb[h * SEQ + j];
    x = x * 0.125f + mask[j];  // 1/sqrt(64) = 0.125
    sv[u] = x;
    mx = fmaxf(mx, x);
  }
  red[t] = mx; __syncthreads();
  for (int off = 64; off > 0; off >>= 1) {
    if (t < off) red[t] = fmaxf(red[t], red[t + off]);
    __syncthreads();
  }
  mx = red[0]; __syncthreads();
  float sum = 0.f;
#pragma unroll
  for (int u = 0; u < 3; ++u) { sv[u] = expf(sv[u] - mx); sum += sv[u]; }
  red[t] = sum; __syncthreads();
  for (int off = 64; off > 0; off >>= 1) {
    if (t < off) red[t] += red[t + off];
    __syncthreads();
  }
  const float inv = 1.0f / red[0];
#pragma unroll
  for (int u = 0; u < 3; ++u) {
    const int j = t + 128 * u;
    probs[((size_t)h * SEQ + i) * SEQ + j] = (bf16_t)(sv[u] * inv);
  }
}

// ---------- Kernel E: ctx = probs @ v -> out[i][h*64+d] ---------------------
__global__ __launch_bounds__(128) void ctx_kernel(
    const bf16_t* __restrict__ probs, const bf16_t* __restrict__ vT,
    float* __restrict__ out) {
  const int it = blockIdx.x;          // 24 i-tiles
  const int h  = blockIdx.y;          // 12 heads
  const int wave = threadIdx.x >> 5;  // 4 waves, ntile = wave (d block)
  const int lane = threadIdx.x & 31;
  const int i0 = it * 16, d0 = wave * 16;
  v8f acc = {};
  for (int ks = 0; ks < 12; ++ks) {
    v16bf a = load_A_bf16(probs + (size_t)h * SEQ * SEQ, SEQ, i0, ks * 32);
    v16bf b = load_B_bf16(vT + (size_t)h * DHEAD * SEQ, SEQ, d0, ks * 32);
    acc = wmma_bf16(a, b, acc);
  }
  float* dst = out + (size_t)i0 * HID + h * DHEAD + d0 + (lane & 15);
  const int mh = (lane & 16) ? 8 : 0;
#pragma unroll
  for (int r = 0; r < 8; ++r) dst[(size_t)(r + mh) * HID] = acc[r];
}

// ---------------------------------------------------------------------------
extern "C" void kernel_launch(void* const* d_in, const int* in_sizes, int n_in,
                              void* d_out, int out_size, void* d_ws, size_t ws_size,
                              hipStream_t stream) {
  const float* hs   = (const float*)d_in[0];
  const float* mask = (const float*)d_in[1];
  const float* pos  = (const float*)d_in[2];
  const float* Wq   = (const float*)d_in[3];
  const float* bq   = (const float*)d_in[4];
  const float* Wk   = (const float*)d_in[5];
  const float* bk   = (const float*)d_in[6];
  const float* Wv   = (const float*)d_in[7];
  const float* bv   = (const float*)d_in[8];
  const float* Wpk  = (const float*)d_in[9];
  // bpk (d_in[10]) contributes a j-independent score offset -> cancels in softmax.
  const float* Wpq  = (const float*)d_in[11];
  const float* bpq  = (const float*)d_in[12];
  float* out = (float*)d_out;

  char* p = (char*)d_ws;
  auto take = [&](size_t bytes) -> void* {
    void* r = (void*)p;
    p += (bytes + 255) & ~(size_t)255;
    return r;
  };
  bf16_t* WT   = (bf16_t*)take((size_t)3 * HID * HID * 2);   // q,k,v transposed
  bf16_t* WPk  = (bf16_t*)take((size_t)HID * HID * 2);
  bf16_t* WPq  = (bf16_t*)take((size_t)HID * HID * 2);
  bf16_t* qb   = (bf16_t*)take((size_t)NHEAD * SEQ * DHEAD * 2);
  bf16_t* kb   = (bf16_t*)take((size_t)NHEAD * SEQ * DHEAD * 2);
  bf16_t* vT   = (bf16_t*)take((size_t)NHEAD * DHEAD * SEQ * 2);
  bf16_t* G    = (bf16_t*)take((size_t)32 * SEQ * HID * 2);
  float*  pbuf = (float*)take((size_t)SEQ * 24 * SEQ * 4);   // c2p + p2c^T merged
  float*  c2c  = (float*)take((size_t)NHEAD * SEQ * SEQ * 4);
  bf16_t* prb  = (bf16_t*)take((size_t)NHEAD * SEQ * SEQ * 2);
  float*  ckb  = (float*)take((size_t)NHEAD * SEQ * 4);

  convert_weights_kernel<<<(HID * HID + 255) / 256, 256, 0, stream>>>(
      Wq, Wk, Wv, Wpk, Wpq, WT, WT + (size_t)HID * HID,
      WT + (size_t)2 * HID * HID, WPk, WPq);
  qkv_kernel<<<dim3(24, 3), 256, 0, stream>>>(hs, WT, bq, bk, bv, qb, kb, vT);
  ck_kernel<<<(NHEAD * SEQ + 255) / 256, 256, 0, stream>>>(kb, bpq, ckb);
  g_kernel<<<dim3(24, 24), 256, 0, stream>>>(qb, kb, WPk, WPq, G);
  pos_kernel<<<SEQ, 256, 0, stream>>>(pos, G, pbuf);
  c2c_kernel<<<dim3(24, NHEAD), 256, 0, stream>>>(qb, kb, c2c);
  softmax_kernel<<<NHEAD * SEQ, 128, 0, stream>>>(c2c, pbuf, ckb, mask, prb);
  ctx_kernel<<<dim3(24, NHEAD), 128, 0, stream>>>(prb, vT, out);
}
